// GP_VAE_7507602833919
// MI455X (gfx1250) — matrix operand order repaired and checked by reading, not compiled, not run on hardware
//
#include <hip/hip_runtime.h>
#include <hip/hip_bf16.h>
#include <stdint.h>

// ---------------------------------------------------------------------------
// Types / constants
// ---------------------------------------------------------------------------
typedef __attribute__((ext_vector_type(16))) __bf16 v16bf;
typedef __attribute__((ext_vector_type(8)))  float  v8f;

#define C_IN  256
#define C_HID 128
#define C_OUT 64
#define NEG_SLOPE 0.2f

static __device__ __forceinline__ unsigned short f2bf(float f) {
  // round-to-nearest-even f32 -> bf16
  unsigned int u = __float_as_uint(f);
  unsigned int r = u + 0x7FFFu + ((u >> 16) & 1u);
  return (unsigned short)(r >> 16);
}

static __device__ __forceinline__ v16bf pack_frag(uint4 a, uint4 b) {
  union { uint4 q[2]; v16bf v; } u;
  u.q[0] = a; u.q[1] = b;
  return u.v;
}

static __device__ __forceinline__ float leaky(float v) {
  return v > 0.0f ? v : NEG_SLOPE * v;
}

// ---------------------------------------------------------------------------
// WMMA GEMM:  C[M x Ntot] = A[M x K](bf16) * B (Bt = [Ntot x K] bf16) + bias
//
// Weights (Bt slice for this block's columns) are staged through LDS in
// KC-wide K chunks; all NT B fragments for a K-step are loaded into distinct
// registers so the WMMAs issue back-to-back behind a single ds wait, while A
// rows stream from global (with a prefetch of the next K chunk).
// Each wave owns one 16-row tile x NT*16 columns. blockIdx.y = column group.
// ---------------------------------------------------------------------------
template<int NT, int KC>
__global__ __launch_bounds__(128) void gemm_bf16_wmma(
    const unsigned short* __restrict__ A,
    const unsigned short* __restrict__ Bt,
    const float* __restrict__ bias,
    float* __restrict__ C,
    int M, int K, int Ntot)
{
  __shared__ unsigned short ldsB[NT * 16 * KC];   // <= 32 KB

  const int lane = threadIdx.x & 31;
  const int wid  = threadIdx.x >> 5;
  const int tile = blockIdx.x * 4 + wid;
  const int row0 = tile * 16;
  const bool active = (row0 < M);
  const int ncol0 = blockIdx.y * (NT * 16);

  const int m   = lane & 15;     // row within tile (A) / column (B, C)
  const int hi  = lane >> 4;     // lane group
  const int kbA = hi * 8;        // A-frag K base within a 32-wide K step
  const int kbB = hi * 8;        // B-frag K base within a 32-wide K step

  const unsigned short* arow = A + (size_t)((active ? row0 : 0) + m) * K;

  v8f acc[NT] = {};

  const int QPR = KC / 8;                 // uint4 quads per LDS row
  for (int kc = 0; kc < K; kc += KC) {
    // ---- cooperative stage of Bt[ncol0 .. ncol0+NT*16) x [kc .. kc+KC) ----
    for (int i = threadIdx.x; i < NT * 16 * QPR; i += 128) {
      int r = i / QPR, q = i % QPR;
      ((uint4*)ldsB)[i] =
          *(const uint4*)(Bt + (size_t)(ncol0 + r) * K + kc + q * 8);
    }
    __syncthreads();

    if (active) {
      // prefetch next A chunk while we compute on this one
      if (kc + KC < K) __builtin_prefetch(arow + kc + KC + kbA, 0, 0);
#pragma unroll
      for (int k0 = 0; k0 < KC; k0 += 32) {
        // A fragment (16x32 bf16): lanes 0-15 -> K {0..7,16..23},
        //                          lanes 16-31 -> K {8..15,24..31}
        uint4 a0 = *(const uint4*)(arow + kc + k0 + kbA);
        uint4 a1 = *(const uint4*)(arow + kc + k0 + kbA + 16);
        v16bf af = pack_frag(a0, a1);

        // Load ALL NT B fragments into distinct registers first so the
        // subsequent WMMAs aren't individually gated on ds round trips.
        v16bf bfr[NT];
#pragma unroll
        for (int t = 0; t < NT; ++t) {
          const unsigned short* brow = ldsB + (t * 16 + m) * KC + k0 + kbB;
          uint4 b0 = *(const uint4*)(brow);
          uint4 b1 = *(const uint4*)(brow + 16);
          bfr[t] = pack_frag(b0, b1);
        }
#pragma unroll
        for (int t = 0; t < NT; ++t) {
          acc[t] = __builtin_amdgcn_wmma_f32_16x16x32_bf16(
              false, af, false, bfr[t], (short)0, acc[t], false, false);
        }
      }
    }
    __syncthreads();   // protect LDS before next chunk restage
  }

  if (!active) return;

  // C/D layout: lane l, VGPR r -> row = r + 8*(l>=16), col = l&15
  const int mmBase = hi * 8;
#pragma unroll
  for (int t = 0; t < NT; ++t) {
    int col = ncol0 + t * 16 + m;
    float b = bias ? bias[col] : 0.0f;
#pragma unroll
    for (int r = 0; r < 8; ++r) {
      C[(size_t)(row0 + mmBase + r) * Ntot + col] = acc[t][r] + b;
    }
  }
}

// ---------------------------------------------------------------------------
// Elementwise helpers
// ---------------------------------------------------------------------------
__global__ void fill_f32(float* __restrict__ p, float v, size_t n) {
  size_t i = (size_t)blockIdx.x * blockDim.x + threadIdx.x;
  if (i < n) p[i] = v;
}

__global__ void conv_bf(const float* __restrict__ in, unsigned short* __restrict__ out,
                        size_t n, int relu) {
  size_t i = (size_t)blockIdx.x * blockDim.x + threadIdx.x;
  if (i >= n) return;
  float f = in[i];
  if (relu) f = fmaxf(f, 0.0f);
  out[i] = f2bf(f);
}

// Wt[n*K + k] = bf16(W[k*Nout + n])   (transpose so B fragments are contiguous)
__global__ void wtrans_bf(const float* __restrict__ W, unsigned short* __restrict__ Wt,
                          int K, int Nout) {
  int i = blockIdx.x * blockDim.x + threadIdx.x;
  if (i >= K * Nout) return;
  int n = i / K, k = i % K;
  Wt[i] = f2bf(W[(size_t)k * Nout + n]);
}

// z = eps*exp(0.5*logvar)+mu ; also emit bf16 copy for the next GAT layer
__global__ void reparam_k(const float* __restrict__ mu, const float* __restrict__ lv,
                          const float* __restrict__ ep, float* __restrict__ z,
                          unsigned short* __restrict__ zbf, size_t n) {
  size_t i = (size_t)blockIdx.x * blockDim.x + threadIdx.x;
  if (i >= n) return;
  float zz = ep[i] * __expf(0.5f * lv[i]) + mu[i];
  z[i] = zz;
  zbf[i] = f2bf(zz);
}

// ---------------------------------------------------------------------------
// GAT pieces
// ---------------------------------------------------------------------------
// per-node dual dot: alpha_src[n] = h[n,:].a_s ; alpha_dst[n] = h[n,:].a_d
__global__ void node_alphas(const float* __restrict__ H, const float* __restrict__ av_s,
                            const float* __restrict__ av_d, float* __restrict__ os,
                            float* __restrict__ od, int Nn, int C) {
  int node = blockIdx.x * 8 + (threadIdx.x >> 5);
  if (node >= Nn) return;
  int lane = threadIdx.x & 31;
  float s = 0.0f, d = 0.0f;
  const float* hrow = H + (size_t)node * C;
  for (int c = lane; c < C; c += 32) {
    float h = hrow[c];
    s += h * av_s[c];
    d += h * av_d[c];
  }
  for (int off = 16; off; off >>= 1) {
    s += __shfl_down(s, off, 32);
    d += __shfl_down(d, off, 32);
  }
  if (lane == 0) { os[node] = s; od[node] = d; }
}

// pass 1: float atomic max of leaky(as[src]+ad[dst]) into m[dst]
__global__ void edge_max(const int* __restrict__ src, const int* __restrict__ dst,
                         int E, int Nn, const float* __restrict__ as_,
                         const float* __restrict__ ad_, float* __restrict__ m) {
  int e = blockIdx.x * blockDim.x + threadIdx.x;
  int Et = E + Nn;
  if (e >= Et) return;
  int s = (e < E) ? src[e] : (e - E);
  int d = (e < E) ? dst[e] : (e - E);
  float v = leaky(as_[s] + ad_[d]);
  if (v >= 0.0f) atomicMax((int*)&m[d], __float_as_int(v));
  else           atomicMin((unsigned int*)&m[d], __float_as_uint(v));
}

// pass 2: p = exp(e - m[dst]); denom[dst] += p
__global__ void edge_exp(const int* __restrict__ src, const int* __restrict__ dst,
                         int E, int Nn, const float* __restrict__ as_,
                         const float* __restrict__ ad_, const float* __restrict__ m,
                         float* __restrict__ p, float* __restrict__ denom) {
  int e = blockIdx.x * blockDim.x + threadIdx.x;
  int Et = E + Nn;
  if (e >= Et) return;
  int s = (e < E) ? src[e] : (e - E);
  int d = (e < E) ? dst[e] : (e - E);
  float v = leaky(as_[s] + ad_[d]);
  float w = __expf(v - m[d]);
  p[e] = w;
  atomicAdd(&denom[d], w);
}

// pass 3: wave per edge; AGG[dst,:] += (p/denom) * H[src,:]
__global__ void edge_scatter(const int* __restrict__ src, const int* __restrict__ dst,
                             int E, int Nn, const float* __restrict__ p,
                             const float* __restrict__ denom, const float* __restrict__ H,
                             float* __restrict__ AGG, int C) {
  int w = (blockIdx.x * blockDim.x + threadIdx.x) >> 5;
  int lane = threadIdx.x & 31;
  int Et = E + Nn;
  if (w >= Et) return;
  int s = (w < E) ? src[w] : (w - E);
  int d = (w < E) ? dst[w] : (w - E);
  float alpha = p[w] / (denom[d] + 1e-16f);
  const float* hs = H + (size_t)s * C;
  float* od = AGG + (size_t)d * C;
  for (int c = lane; c < C; c += 32) atomicAdd(&od[c], alpha * hs[c]);
}

// ---------------------------------------------------------------------------
// Host-side launch helpers
// ---------------------------------------------------------------------------
static inline int cdiv(long long a, long long b) { return (int)((a + b - 1) / b); }

static void launch_gemm(const unsigned short* A, const unsigned short* Bt,
                        const float* bias, float* C, int M, int K, int Ntot,
                        hipStream_t s) {
  dim3 blk(128);
  dim3 grd(cdiv((M + 15) / 16, 4), 1);
  if (Ntot == 64) {                 // fc layers: K=128
    gemm_bf16_wmma<4, 128><<<grd, blk, 0, s>>>(A, Bt, bias, C, M, K, Ntot);
  } else if (Ntot == 128) {
    if (K == 64) {                  // gat3
      gemm_bf16_wmma<8, 64><<<grd, blk, 0, s>>>(A, Bt, bias, C, M, K, Ntot);
    } else {                        // gat1 (K=256, 2 chunks), gat2 (K=128)
      gemm_bf16_wmma<8, 128><<<grd, blk, 0, s>>>(A, Bt, bias, C, M, K, Ntot);
    }
  } else {                          // gat4: Ntot=256, K=128
    grd.y = Ntot / 128;
    gemm_bf16_wmma<8, 128><<<grd, blk, 0, s>>>(A, Bt, bias, C, M, K, Ntot);
  }
}

// ---------------------------------------------------------------------------
// Entry point
// ---------------------------------------------------------------------------
extern "C" void kernel_launch(void* const* d_in, const int* in_sizes, int n_in,
                              void* d_out, int out_size, void* d_ws, size_t ws_size,
                              hipStream_t stream) {
  const float* x   = (const float*)d_in[0];
  const int*   ei  = (const int*)  d_in[1];
  const float* W1  = (const float*)d_in[2];
  const float* a1s = (const float*)d_in[3];
  const float* a1d = (const float*)d_in[4];
  const float* W2  = (const float*)d_in[5];
  const float* a2s = (const float*)d_in[6];
  const float* a2d = (const float*)d_in[7];
  const float* Wmu = (const float*)d_in[8];
  const float* bmu = (const float*)d_in[9];
  const float* Wlv = (const float*)d_in[10];
  const float* blv = (const float*)d_in[11];
  const float* W3  = (const float*)d_in[12];
  const float* a3s = (const float*)d_in[13];
  const float* a3d = (const float*)d_in[14];
  const float* W4  = (const float*)d_in[15];
  const float* a4s = (const float*)d_in[16];
  const float* a4d = (const float*)d_in[17];
  const float* eps = (const float*)d_in[18];

  const int Nn = in_sizes[0] / C_IN;   // 50000
  const int E  = in_sizes[1] / 2;      // 800000
  const int Et = E + Nn;
  const int* srcP = ei;
  const int* dstP = ei + E;

  // ---- carve workspace ----
  uint8_t* ws = (uint8_t*)d_ws;
  size_t off = 0;
  auto carve = [&](size_t bytes) -> uint8_t* {
    off = (off + 255) & ~(size_t)255;
    uint8_t* p = ws + off;
    off += bytes;
    return p;
  };
  float*          H    = (float*)carve((size_t)Nn * C_IN * 4);   // h = x@W (max 256 wide)
  float*          AGG  = (float*)carve((size_t)Nn * C_HID * 4);  // aggregated output (<=128 wide)
  unsigned short* Xbf  = (unsigned short*)carve((size_t)Nn * C_IN * 2);
  float*          asrc = (float*)carve((size_t)Nn * 4);
  float*          adst = (float*)carve((size_t)Nn * 4);
  float*          mmax = (float*)carve((size_t)Nn * 4);
  float*          dnm  = (float*)carve((size_t)Nn * 4);
  float*          pbuf = (float*)carve((size_t)Et * 4);
  unsigned short* W1t  = (unsigned short*)carve((size_t)C_IN  * C_HID * 2);
  unsigned short* W2t  = (unsigned short*)carve((size_t)C_HID * C_HID * 2);
  unsigned short* Wmut = (unsigned short*)carve((size_t)C_HID * C_OUT * 2);
  unsigned short* Wlvt = (unsigned short*)carve((size_t)C_HID * C_OUT * 2);
  unsigned short* W3t  = (unsigned short*)carve((size_t)C_OUT * C_HID * 2);
  unsigned short* W4t  = (unsigned short*)carve((size_t)C_HID * C_IN  * 2);

  float* recon = (float*)d_out;
  float* muO   = recon + (size_t)Nn * C_IN;
  float* lvO   = muO + (size_t)Nn * C_OUT;
  float* zO    = lvO + (size_t)Nn * C_OUT;

  const float NEG_INF = -__builtin_inff();

  // ---- weight conversion (transposed bf16) ----
  wtrans_bf<<<cdiv(C_IN  * C_HID, 256), 256, 0, stream>>>(W1,  W1t,  C_IN,  C_HID);
  wtrans_bf<<<cdiv(C_HID * C_HID, 256), 256, 0, stream>>>(W2,  W2t,  C_HID, C_HID);
  wtrans_bf<<<cdiv(C_HID * C_OUT, 256), 256, 0, stream>>>(Wmu, Wmut, C_HID, C_OUT);
  wtrans_bf<<<cdiv(C_HID * C_OUT, 256), 256, 0, stream>>>(Wlv, Wlvt, C_HID, C_OUT);
  wtrans_bf<<<cdiv(C_OUT * C_HID, 256), 256, 0, stream>>>(W3,  W3t,  C_OUT, C_HID);
  wtrans_bf<<<cdiv(C_HID * C_IN , 256), 256, 0, stream>>>(W4,  W4t,  C_HID, C_IN);

  // ---- one GAT layer ----
  auto gat = [&](const unsigned short* Xb, int K, int Cout, const unsigned short* Wt,
                 const float* av_s, const float* av_d, float* aggOut) {
    launch_gemm(Xb, Wt, nullptr, H, Nn, K, Cout, stream);
    node_alphas<<<cdiv(Nn, 8), 256, 0, stream>>>(H, av_s, av_d, asrc, adst, Nn, Cout);
    fill_f32<<<cdiv(Nn, 256), 256, 0, stream>>>(mmax, NEG_INF, (size_t)Nn);
    fill_f32<<<cdiv(Nn, 256), 256, 0, stream>>>(dnm, 0.0f, (size_t)Nn);
    fill_f32<<<cdiv((long long)Nn * Cout, 256), 256, 0, stream>>>(aggOut, 0.0f, (size_t)Nn * Cout);
    edge_max<<<cdiv(Et, 256), 256, 0, stream>>>(srcP, dstP, E, Nn, asrc, adst, mmax);
    edge_exp<<<cdiv(Et, 256), 256, 0, stream>>>(srcP, dstP, E, Nn, asrc, adst, mmax, pbuf, dnm);
    edge_scatter<<<cdiv((long long)Et * 32, 256), 256, 0, stream>>>(
        srcP, dstP, E, Nn, pbuf, dnm, H, aggOut, Cout);
  };

  // ---- encode ----
  conv_bf<<<cdiv((long long)Nn * C_IN, 256), 256, 0, stream>>>(x, Xbf, (size_t)Nn * C_IN, 0);
  gat(Xbf, C_IN, C_HID, W1t, a1s, a1d, AGG);                       // gat1
  conv_bf<<<cdiv((long long)Nn * C_HID, 256), 256, 0, stream>>>(AGG, Xbf, (size_t)Nn * C_HID, 1); // relu
  gat(Xbf, C_HID, C_HID, W2t, a2s, a2d, AGG);                      // gat2

  // ---- fc_mu / fc_logvar / reparam ----
  conv_bf<<<cdiv((long long)Nn * C_HID, 256), 256, 0, stream>>>(AGG, Xbf, (size_t)Nn * C_HID, 0);
  launch_gemm(Xbf, Wmut, bmu, muO, Nn, C_HID, C_OUT, stream);
  launch_gemm(Xbf, Wlvt, blv, lvO, Nn, C_HID, C_OUT, stream);
  reparam_k<<<cdiv((long long)Nn * C_OUT, 256), 256, 0, stream>>>(
      muO, lvO, eps, zO, Xbf, (size_t)Nn * C_OUT);

  // ---- decode ----
  gat(Xbf, C_OUT, C_HID, W3t, a3s, a3d, AGG);                      // gat3
  conv_bf<<<cdiv((long long)Nn * C_HID, 256), 256, 0, stream>>>(AGG, Xbf, (size_t)Nn * C_HID, 1); // relu
  gat(Xbf, C_HID, C_IN, W4t, a4s, a4d, recon);                     // gat4 -> recon (in d_out)
}
